// LSTM_noforget2_25305947308387
// MI455X (gfx1250) — matrix operand order repaired and checked by reading, not compiled, not run on hardware
//
#include <hip/hip_runtime.h>
#include <hip/hip_bf16.h>

// ---------------------------------------------------------------------------
// LSTM (no forget gate), BS=1, SEQ=4096, HIDDEN=2048, COMB=2560.
// MI455X strategy:
//   * bf16 weights (31.5 MB) live in the 192 MB L2; streamed every step
//     through v_wmma_f32_16x16x32_bf16 (fp32 accumulate).
//   * Persistent kernel: 32 blocks x 128 threads = 128 wave32 waves; wave w
//     owns gate rows [16w,16w+16) for all 3 gates -> wave-local nonlinearity,
//     one grid barrier per timestep (monotonic atomic + s_sleep backoff).
//   * comb staged in LDS; A fragments (row 0 = comb chunk) built branch-free:
//     inactive lanes read a zeroed LDS pad (no exec-mask dance, no WMMA->VALU
//     hazard NOPs).
//   * One-chunk-deep software pipeline: chunk c+1 weight/A loads issued while
//     chunk c's WMMAs execute.
// ---------------------------------------------------------------------------

typedef __attribute__((ext_vector_type(16))) __bf16        v16bf;
typedef __attribute__((ext_vector_type(8)))  float         v8f;
typedef __attribute__((ext_vector_type(8)))  unsigned int  v8u;

#define INPUT_D   512
#define HIDDEN_D  2048
#define OUTPUT_D  512
#define SEQ_T     4096
#define COMB_D    2560
#define KCH       80            // COMB_D / 32
#define NWAVES    128           // one wave per 16 gate rows
#define NBLK      32
#define TPB       128           // 4 waves/block (wave32)

// packed weights: [gate(3)][wave(128)][chunk(80)][lane(32)][elem(16)] bf16
#define PACKW_HALVES (3u * NWAVES * KCH * 32u * 16u)   // 15,728,640 halves
#define HB_OFF   ((size_t)PACKW_HALVES * 2u)           // 31,457,280 B
#define HF_OFF   (HB_OFF + 2u * HIDDEN_D * 2u)         // + 8 KB
#define BAR_OFF  (HF_OFF + (size_t)HIDDEN_D * 4u)      // + 8 KB

#define ZPAD_B   5120           // byte offset of 64B zero pad in LDS

__device__ __forceinline__ unsigned short f2bf(float f) {
  unsigned u = __float_as_uint(f);
  unsigned r = u + 0x7FFFu + ((u >> 16) & 1u);   // round-to-nearest-even
  return (unsigned short)(r >> 16);
}

// ---------------- prologue: zero h0 / barrier ----------------
__global__ void init_ws(unsigned short* hb, unsigned* bar) {
  int i = blockIdx.x * blockDim.x + threadIdx.x;
  if (i < 2 * HIDDEN_D) hb[i] = 0;
  if (i == 0) *bar = 0u;
}

// ---------------- prologue: swizzle weights fp32 -> bf16 B-layout ----------
__global__ void pack_w(const float* __restrict__ Wi, const float* __restrict__ Wg,
                       const float* __restrict__ Wo, unsigned short* __restrict__ packw) {
  unsigned e = blockIdx.x * blockDim.x + threadIdx.x;
  if (e >= PACKW_HALVES) return;
  unsigned n    = (e >> 4) & 15u;   // N = lane%16 (column of B == gate row within tile)
  unsigned l    = (e >> 4) & 31u;   // lane; K = table order over (lane/16, elem)
  unsigned rest = e >> 9;           // (g*128 + w)*80 + c
  unsigned c    = rest % KCH;
  unsigned gw   = rest / KCH;
  unsigned w    = gw & 127u;
  unsigned g    = gw >> 7;
  const float* W = (g == 0) ? Wi : (g == 1) ? Wg : Wo;
  unsigned row = w * 16u + n;
  unsigned k   = c * 32u + (e & 7u) + 8u * (l >> 4) + 16u * ((e >> 3) & 1u);
  packw[e] = f2bf(W[(size_t)row * COMB_D + k]);
}

// ---------------- persistent recurrent kernel ----------------
__global__ __launch_bounds__(TPB, 1)
void lstm_main(const unsigned short* __restrict__ packw,
               const float* __restrict__ x,       // [SEQ, 512] fp32
               const float* __restrict__ bi, const float* __restrict__ bg,
               const float* __restrict__ bo,
               const float* __restrict__ Wout, const float* __restrict__ bout,
               const float* __restrict__ hidden_in,
               unsigned short* __restrict__ hb,   // [2][2048] bf16 double buffer
               float* __restrict__ hf,            // [2048] final h fp32
               unsigned* __restrict__ bar,
               float* __restrict__ out)           // [512 out | 2048 hidden]
{
  // 5120 B comb staging + 64 B zero pad (A source for inactive lanes +
  // safe landing zone for the one-iteration pipeline overrun)
  __shared__ uint4 combS4[COMB_D / 8 + 4];        // 5184 B
  unsigned short* combS = (unsigned short*)combS4;
  char*           combB = (char*)combS4;

  const int tid  = threadIdx.x;
  const int lane = tid & 31;
  const int wv   = blockIdx.x * (TPB / 32) + (tid >> 5);  // 0..127
  const int rowbase = wv * 16;

  // zero the pad once (ordered before use by the in-loop __syncthreads)
  if (tid < 16) ((unsigned*)(combB + ZPAD_B))[tid] = 0u;

  // branch-free A addressing: active lanes (0,16) walk the comb vector,
  // inactive lanes stay parked on the zero pad.
  const bool active = (lane & 15) == 0;
  const int  aBase  = active ? ((lane >> 4) << 4) : ZPAD_B;  // lane16 -> +16B
  const int  aStep  = active ? 64 : 0;

  const unsigned short* pWi = packw + (size_t)(0 * NWAVES + wv) * (KCH * 512u) + lane * 16u;
  const unsigned short* pWg = packw + (size_t)(1 * NWAVES + wv) * (KCH * 512u) + lane * 16u;
  const unsigned short* pWo = packw + (size_t)(2 * NWAVES + wv) * (KCH * 512u) + lane * 16u;

  for (int t = 0; t < SEQ_T; ++t) {
    const int rd = t & 1;                         // read buffer index
    // ---- stage comb = [x_t | h] into LDS as bf16 ----
    for (int i = tid; i < INPUT_D; i += TPB)
      combS[i] = f2bf(x[(size_t)t * INPUT_D + i]);
    {
      const uint4* hsrc = (const uint4*)(hb + rd * HIDDEN_D);
      uint4* dst = (uint4*)(combS + INPUT_D);
      for (int i = tid; i < HIDDEN_D / 8; i += TPB) dst[i] = hsrc[i];
    }
    __syncthreads();

    // ---- three gate GEMVs via WMMA bf16, fp32 accumulate, 1-deep pipeline --
    v8f ai = {}; v8f ag = {}; v8f ao = {};
    int aAddr = aBase;

    v16bf Bi = *(const v16bf*)(pWi);
    v16bf Bg = *(const v16bf*)(pWg);
    v16bf Bo = *(const v16bf*)(pWo);
    uint4 lo = *(const uint4*)(combB + aAddr);
    uint4 hi = *(const uint4*)(combB + aAddr + 32);
    aAddr += aStep;

    #pragma unroll 2
    for (int c = 0; c < KCH; ++c) {
      // assemble A for chunk c (lane0: K0..7,K16..23; lane16: K8..15,K24..31)
      v8u av;
      av[0] = lo.x; av[1] = lo.y; av[2] = lo.z; av[3] = lo.w;
      av[4] = hi.x; av[5] = hi.y; av[6] = hi.z; av[7] = hi.w;
      const v16bf a = __builtin_bit_cast(v16bf, av);

      // issue chunk c+1 loads (clamped for B; A overrun lands in zero pad)
      const int cn = (c + 1 < KCH) ? c + 1 : c;
      const v16bf nBi = *(const v16bf*)(pWi + cn * 512);
      const v16bf nBg = *(const v16bf*)(pWg + cn * 512);
      const v16bf nBo = *(const v16bf*)(pWo + cn * 512);
      const uint4 nlo = *(const uint4*)(combB + aAddr);
      const uint4 nhi = *(const uint4*)(combB + aAddr + 32);
      aAddr += aStep;

      ai = __builtin_amdgcn_wmma_f32_16x16x32_bf16(false, a, false, Bi, (short)0, ai, false, false);
      ag = __builtin_amdgcn_wmma_f32_16x16x32_bf16(false, a, false, Bg, (short)0, ag, false, false);
      ao = __builtin_amdgcn_wmma_f32_16x16x32_bf16(false, a, false, Bo, (short)0, ao, false, false);

      Bi = nBi; Bg = nBg; Bo = nBo; lo = nlo; hi = nhi;
    }
    __syncthreads();   // done reading combS before next step refills it

    // ---- nonlinearity: D row 0 is VGPR0 (elem 0) across lanes 0..15 ----
    if (lane < 16) {
      const int r = rowbase + lane;
      const float zi = ai[0] + bi[r];
      const float zg = ag[0] + bg[r];
      const float zo = ao[0] + bo[r];
      const float it = 1.f / (1.f + __expf(-zi));
      const float gt = tanhf(zg);
      const float ot = 1.f / (1.f + __expf(-zo));
      const float hn = ot * tanhf(it * gt);
      hb[(1 - rd) * HIDDEN_D + r] = f2bf(hn);
      if (t == SEQ_T - 1) hf[r] = hn;
    }

    // ---- device-wide barrier (monotonic counter, release/acquire) ----
    __threadfence();
    __syncthreads();
    if (tid == 0) {
      __hip_atomic_fetch_add(bar, 1u, __ATOMIC_ACQ_REL, __HIP_MEMORY_SCOPE_AGENT);
      const unsigned tgt = (unsigned)NBLK * (unsigned)(t + 1);
      while (__hip_atomic_load(bar, __ATOMIC_ACQUIRE, __HIP_MEMORY_SCOPE_AGENT) < tgt)
        __builtin_amdgcn_s_sleep(2);
    }
    __syncthreads();
  }

  // ---- output = Wout @ hT + bout (fp32, one-time) ----
  for (int oo = 0; oo < OUTPUT_D / NWAVES; ++oo) {   // 4 rows per wave
    const int o = wv * (OUTPUT_D / NWAVES) + oo;
    float p = 0.f;
    for (int j = lane; j < HIDDEN_D; j += 32)
      p += Wout[(size_t)o * HIDDEN_D + j] * hf[j];
    for (int off = 16; off; off >>= 1) p += __shfl_xor(p, off, 32);
    if (lane == 0) out[o] = p + bout[o];
  }
  // ---- passthrough `hidden` tensor (second tuple element) ----
  for (int i = blockIdx.x * TPB + tid; i < HIDDEN_D; i += NBLK * TPB)
    out[OUTPUT_D + i] = hidden_in[i];
}

// ---------------------------------------------------------------------------
extern "C" void kernel_launch(void* const* d_in, const int* in_sizes, int n_in,
                              void* d_out, int out_size, void* d_ws, size_t ws_size,
                              hipStream_t stream) {
  (void)in_sizes; (void)n_in; (void)out_size; (void)ws_size;
  const float* x      = (const float*)d_in[0];
  const float* hidden = (const float*)d_in[1];
  const float* Wi     = (const float*)d_in[2];
  const float* bi     = (const float*)d_in[3];
  const float* Wg     = (const float*)d_in[4];
  const float* bg     = (const float*)d_in[5];
  const float* Wo     = (const float*)d_in[6];
  const float* bo     = (const float*)d_in[7];
  const float* Wout   = (const float*)d_in[8];
  const float* bout   = (const float*)d_in[9];
  float* out = (float*)d_out;

  char* ws = (char*)d_ws;
  unsigned short* packw = (unsigned short*)(ws);
  unsigned short* hb    = (unsigned short*)(ws + HB_OFF);
  float*          hf    = (float*)(ws + HF_OFF);
  unsigned*       bar   = (unsigned*)(ws + BAR_OFF);

  init_ws<<<(2 * HIDDEN_D + 255) / 256, 256, 0, stream>>>(hb, bar);
  pack_w<<<(PACKW_HALVES + 255) / 256, 256, 0, stream>>>(Wi, Wg, Wo, packw);
  lstm_main<<<NBLK, TPB, 0, stream>>>(packw, x, bi, bg, bo, Wout, bout,
                                      hidden, hb, hf, bar, out);
}
